// BidirRWKV6MultiScaleTimeMix_57844619542619
// MI455X (gfx1250) — compile-verified, matrix operand order checked by reading, not run on hardware
//
#include <hip/hip_runtime.h>
#include <hip/hip_bf16.h>

// ---------------- types ----------------
typedef __attribute__((ext_vector_type(16))) __bf16 v16bf;
typedef __attribute__((ext_vector_type(8)))  float  v8f;
typedef __attribute__((ext_vector_type(4)))  unsigned int v4u;
typedef __attribute__((ext_vector_type(8)))  int    v8i;
typedef __attribute__((ext_vector_type(4)))  int    v4i;

static constexpr int Bc  = 2;
static constexpr int Tc  = 2048;
static constexpr int Dm  = 1024;
static constexpr int Hc  = 16;
static constexpr int Kc  = 64;
static constexpr int BT  = Bc * Tc;          // 4096
static constexpr float EPSc = 1e-5f * 64.0f; // eps * head_size_divisor^2

__device__ __forceinline__ __bf16 f2bf(float f) {
    unsigned int u = __builtin_bit_cast(unsigned int, f);
    unsigned int r = (u + 0x7FFFu + ((u >> 16) & 1u)) >> 16;
    unsigned short s = (unsigned short)r;
    return __builtin_bit_cast(__bf16, s);
}

// Load a 16x32 bf16 fragment (A-layout, also used for B with pre-transposed
// storage [n][k]): lane<16 -> row=lane, cols {0..7,16..23}; lane>=16 ->
// row=lane-16, cols {8..15,24..31}. stride in elements.
__device__ __forceinline__ v16bf load_frag16(const __bf16* base, int stride, int lane) {
    const int row = lane & 15;
    const int cb  = (lane >> 4) << 3;
    const __bf16* p = base + (size_t)row * stride + cb;
    v16bf f;
#pragma unroll
    for (int i = 0; i < 8; ++i) f[i] = p[i];
#pragma unroll
    for (int i = 0; i < 8; ++i) f[i + 8] = p[i + 16];
    return f;
}

// ---------------- TDM (Tensor Data Mover) helpers ----------------
// 2-D tile load global->LDS via tensor_load_to_lds. Tile is tile_w x tile_h
// bf16 elements, row stride in global memory = stride_elems, written
// contiguously (row-major, tile_w wide) at the LDS destination.
// Tensor dims are set equal to tile dims (tiles never cross real bounds here),
// so no OOB clamping paths trigger. This toolchain's builtin takes 6 args:
// (g0 u32x4, g1 i32x8, g2 i32x4, g3 i32x4, extra i32x8, cpol) — groups 2/3
// and the extra group are for >2-D / gather mode, zero for plain 2-D tiles.
__device__ __forceinline__ void tdm_load_2d(const __bf16* g, __bf16* lds,
                                            int tile_w, int tile_h,
                                            int stride_elems, int lane) {
#if defined(__HIP_DEVICE_COMPILE__) && __has_builtin(__builtin_amdgcn_tensor_load_to_lds)
    unsigned long long ga = (unsigned long long)(size_t)g;
    unsigned lds_off = (unsigned)(size_t)lds;   // LDS aperture keeps offset in addr[31:0]
    v4u g0;
    g0[0] = 1u;                                  // count=1, user-mode descriptor
    g0[1] = lds_off;                             // lds_addr (bytes)
    g0[2] = (unsigned)(ga & 0xFFFFFFFFu);        // global_addr[31:0]
    g0[3] = (unsigned)((ga >> 32) & 0x1FFFFFFu)  // global_addr[56:32]
          | (2u << 30);                          // type = 2 ("image")
    unsigned td0 = (unsigned)tile_w;             // tensor_dim0 == tile_dim0
    unsigned td1 = (unsigned)tile_h;             // tensor_dim1 == tile_dim1
    unsigned long long st = (unsigned long long)stride_elems;
    v8i g1;
    g1[0] = (int)(1u << 16);                     // data_size=1 (2 bytes/elem)
    g1[1] = (int)((td0 & 0xFFFFu) << 16);        // tensor_dim0[15:0] @ bits 63:48
    g1[2] = (int)((td0 >> 16) | ((td1 & 0xFFFFu) << 16)); // dim0 hi / dim1 lo
    g1[3] = (int)((td1 >> 16) | ((unsigned)tile_w << 16)); // dim1 hi / tile_dim0
    g1[4] = (int)((unsigned)tile_h & 0xFFFFu);   // tile_dim1 (tile_dim2 = 0)
    g1[5] = (int)(st & 0xFFFFFFFFull);           // tensor_dim0_stride[31:0]
    g1[6] = (int)((st >> 32) & 0xFFFFull);       // stride[47:32]; dim1_stride lo = 0
    g1[7] = 0;
    v4i g2 = {0, 0, 0, 0};
    v4i g3 = {0, 0, 0, 0};
    v8i g4 = {0, 0, 0, 0, 0, 0, 0, 0};
    __builtin_amdgcn_tensor_load_to_lds(g0, g1, g2, g3, g4, 0);
    (void)lane;
#else
    // Fallback: cooperative copy by the issuing wave (32 lanes).
    const int total = tile_w * tile_h;
    for (int e = lane * 8; e < total; e += 32 * 8) {
        int r = e / tile_w, c = e % tile_w;
#pragma unroll
        for (int q = 0; q < 8; ++q)
            lds[e + q] = g[(size_t)r * stride_elems + c + q];
    }
#endif
}

template <int N>
__device__ __forceinline__ void tdm_wait() {
#if defined(__HIP_DEVICE_COMPILE__) && __has_builtin(__builtin_amdgcn_s_wait_tensorcnt)
    __builtin_amdgcn_s_wait_tensorcnt((short)N);
#endif
}

// ---------------- kernel 1: token shift + LoRA mixes + decay row ----------------
__global__ void rwkv_mix_kernel(const float* __restrict__ x,
                                const float* __restrict__ tmx, const float* __restrict__ tmw,
                                const float* __restrict__ tmk, const float* __restrict__ tmv,
                                const float* __restrict__ tmr, const float* __restrict__ tmg,
                                const float* __restrict__ w1,  const float* __restrict__ w2,
                                const float* __restrict__ tdecay,
                                const float* __restrict__ dw1, const float* __restrict__ dw2,
                                const float* __restrict__ hdb,
                                __bf16* __restrict__ xk_bf, __bf16* __restrict__ xv_bf,
                                __bf16* __restrict__ xr_bf, __bf16* __restrict__ xg_bf,
                                float* __restrict__ wrow) {
    __shared__ float xs[Dm], dxp[Dm], xwb[Dm];
    __shared__ float hb[160], hw[64], hsum[16];
    const int id  = blockIdx.x;          // 0..BT-1
    const int t   = id % Tc;
    const int tid = threadIdx.x;

    for (int d = tid; d < Dm; d += 256) {
        float xc = x[(size_t)id * Dm + d];
        float xp = (t > 0)      ? x[(size_t)(id - 1) * Dm + d] : 0.f;
        float xn = (t < Tc - 1) ? x[(size_t)(id + 1) * Dm + d] : 0.f;
        xs[d]  = xc;
        dxp[d] = 0.5f * (xp + xn) - xc;
    }
    if (tid < 16) hsum[tid] = 0.f;
    __syncthreads();

    // h = tanh(xxx @ w1), w1: [1024,160]
    for (int j = tid; j < 160; j += 256) {
        float s = 0.f;
        for (int d = 0; d < Dm; ++d)
            s += (xs[d] + dxp[d] * tmx[d]) * w1[(size_t)d * 160 + j];
        hb[j] = tanhf(s);
    }
    __syncthreads();

    // per-d branch mixes; w2: [5,32,1024]
    for (int d = tid; d < Dm; d += 256) {
        float m[5];
#pragma unroll
        for (int f = 0; f < 5; ++f) {
            float s = 0.f;
            for (int c = 0; c < 32; ++c)
                s += hb[f * 32 + c] * w2[(size_t)(f * 32 + c) * Dm + d];
            m[f] = s;
        }
        float xc = xs[d], dd = dxp[d];
        size_t o = (size_t)id * Dm + d;
        xwb[d]   = xc + dd * (tmw[d] + m[0]);
        xk_bf[o] = f2bf(xc + dd * (tmk[d] + m[1]));
        xv_bf[o] = f2bf(xc + dd * (tmv[d] + m[2]));
        xr_bf[o] = f2bf(xc + dd * (tmr[d] + m[3]));
        xg_bf[o] = f2bf(xc + dd * (tmg[d] + m[4]));
    }
    __syncthreads();

    // hw = tanh(xw @ dw1), dw1: [1024,64]
    for (int j = tid; j < 64; j += 256) {
        float s = 0.f;
        for (int d = 0; d < Dm; ++d) s += xwb[d] * dw1[(size_t)d * 64 + j];
        hw[j] = tanhf(s);
    }
    __syncthreads();

    // w_d = tdecay + hw @ dw2; head mean of -exp(w)
    for (int d = tid; d < Dm; d += 256) {
        float s = tdecay[d];
        for (int j = 0; j < 64; ++j) s += hw[j] * dw2[(size_t)j * Dm + d];
        atomicAdd(&hsum[d >> 6], -expf(s));
    }
    __syncthreads();
    if (tid < 16)
        wrow[(size_t)id * Hc + tid] = hsum[tid] * (1.0f / 64.0f) * expf(hdb[tid]);
}

// ---------------- kernel 2: per-(b,h) cumsum of decay row ----------------
__global__ void rwkv_cumsum_kernel(const float* __restrict__ wrow, float* __restrict__ C) {
    if (threadIdx.x != 0) return;
    int bh = blockIdx.x;            // 0..31
    int b = bh / Hc, h = bh % Hc;
    float c = 0.f;
    for (int t = 0; t < Tc; ++t) {
        c += wrow[(size_t)(b * Tc + t) * Hc + h];
        C[(size_t)bh * Tc + t] = c;
    }
}

// ---------------- kernel 3: weight convert + transpose ----------------
__global__ void rwkv_wtrans_kernel(const float* __restrict__ W, __bf16* __restrict__ Wt,
                                   int In, int Out) {
    int gid = blockIdx.x * blockDim.x + threadIdx.x;
    if (gid >= In * Out) return;
    int k = gid / Out, n = gid % Out;
    Wt[(size_t)n * In + k] = f2bf(W[gid]);
}

// ---------------- kernel 4: WMMA GEMM with TDM double-buffered LDS tiles ----
// out[M,N] = A[M,K] * Bt[N,K]^T.  Workgroup = 4 waves, 64x64 output tile.
// Wave 0 drives the Tensor Data Mover: per 32-wide k-step it DMAs the next
// A(64x32) and Bt(64x32) bf16 tiles into the spare LDS buffer, waits
// tensorcnt for the current ones, then all 4 waves run WMMAs out of LDS.
// mode 0: bf16 out (scaled); mode 1: f32 silu(out); mode 2: f32 out
__global__ void rwkv_gemm_kernel(const __bf16* __restrict__ A, const __bf16* __restrict__ Bt,
                                 void* __restrict__ out, int M, int N, int Kd,
                                 int mode, float scale) {
    __shared__ __bf16 Abuf[2][64 * 32];
    __shared__ __bf16 Bbuf[2][64 * 32];
    const int lane = threadIdx.x & 31;
    const int wv   = threadIdx.x >> 5;
    const int mt   = blockIdx.x, nt = blockIdx.y;

    const __bf16* Abase = A  + (size_t)(mt * 64) * Kd;
    const __bf16* Bbase = Bt + (size_t)(nt * 64) * Kd;

    v8f acc[4];
#pragma unroll
    for (int j = 0; j < 4; ++j)
#pragma unroll
        for (int i = 0; i < 8; ++i) acc[j][i] = 0.f;

    if (wv == 0) {
        tdm_load_2d(Abase, &Abuf[0][0], 32, 64, Kd, lane);
        tdm_load_2d(Bbase, &Bbuf[0][0], 32, 64, Kd, lane);
    }

    for (int k0 = 0; k0 < Kd; k0 += 32) {
        const int  cur  = (k0 >> 5) & 1;
        const bool more = (k0 + 32) < Kd;
        if (wv == 0) {
            if (more) {
                tdm_load_2d(Abase + k0 + 32, &Abuf[cur ^ 1][0], 32, 64, Kd, lane);
                tdm_load_2d(Bbase + k0 + 32, &Bbuf[cur ^ 1][0], 32, 64, Kd, lane);
                tdm_wait<2>();   // current tiles landed; next 2 still in flight
            } else {
                tdm_wait<0>();
            }
        }
        __syncthreads();
        v16bf a = load_frag16(&Abuf[cur][wv * 16 * 32], 32, lane);
#pragma unroll
        for (int j = 0; j < 4; ++j) {
            v16bf b = load_frag16(&Bbuf[cur][j * 16 * 32], 32, lane);
            acc[j] = __builtin_amdgcn_wmma_f32_16x16x32_bf16(false, a, false, b,
                                                             (short)0, acc[j], false, false);
        }
        __syncthreads();
    }

    const int mrow = mt * 64 + wv * 16 + ((lane >> 4) << 3);
#pragma unroll
    for (int j = 0; j < 4; ++j) {
        const int n = nt * 64 + j * 16 + (lane & 15);
#pragma unroll
        for (int i = 0; i < 8; ++i) {
            size_t o = (size_t)(mrow + i) * N + n;
            float v = acc[j][i] * scale;
            if (mode == 0)      ((__bf16*)out)[o] = f2bf(v);
            else if (mode == 1) ((float*)out)[o]  = v / (1.f + expf(-v));  // silu
            else                ((float*)out)[o]  = v;
        }
    }
}

// ---------------- kernel 5: V transpose to [b,h,k,t] ----------------
__global__ void rwkv_vtrans_kernel(const __bf16* __restrict__ v, __bf16* __restrict__ vt) {
    int gid = blockIdx.x * blockDim.x + threadIdx.x;
    if (gid >= BT * Dm) return;
    int tg = gid >> 10, d = gid & 1023;
    int b = tg / Tc, t = tg % Tc;
    int h = d >> 6, kk = d & 63;
    vt[((size_t)(b * Hc + h) * Kc + kk) * Tc + t] = v[gid];
}

// ---------------- kernel 6: fused LION bidirectional decay attention ----------------
__global__ void rwkv_attn_kernel(const __bf16* __restrict__ rb, const __bf16* __restrict__ kb,
                                 const __bf16* __restrict__ vt, const float* __restrict__ C,
                                 const float* __restrict__ slow_scale,
                                 const float* __restrict__ decay_mix,
                                 float* __restrict__ y) {
    __shared__ __bf16 smem[4][2][16 * 32];
    const int lane = threadIdx.x & 31;
    const int wv   = threadIdx.x >> 5;
    const int qt   = blockIdx.x * 4 + wv;        // q tile (of 16)
    const int h    = blockIdx.y;
    const int b    = blockIdx.z;

    const __bf16* rbase = rb + (size_t)(b * Tc) * Dm + h * Kc;
    const __bf16* kbase = kb + (size_t)(b * Tc) * Dm + h * Kc;
    const __bf16* vbase = vt + (size_t)(b * Hc + h) * Kc * Tc;
    const float*  Cb    = C  + (size_t)(b * Hc + h) * Tc;

    const float sh = 1.f / (1.f + expf(-slow_scale[h]));

    // preload r fragments for this q tile (k = 0..31, 32..63)
    v16bf rA0 = load_frag16(rbase + (size_t)(qt * 16) * Dm,      Dm, lane);
    v16bf rA1 = load_frag16(rbase + (size_t)(qt * 16) * Dm + 32, Dm, lane);

    float Cq[8];
#pragma unroll
    for (int i = 0; i < 8; ++i) Cq[i] = Cb[qt * 16 + ((lane >> 4) << 3) + i];

    v8f yf[4], ys[4];
#pragma unroll
    for (int j = 0; j < 4; ++j)
#pragma unroll
        for (int i = 0; i < 8; ++i) { yf[j][i] = 0.f; ys[j][i] = 0.f; }

    for (int sb = 0; sb < Tc; sb += 32) {
        // S tiles: [16q x 16s] each, contraction over head dim (64)
        v8f S0, S1;
#pragma unroll
        for (int i = 0; i < 8; ++i) { S0[i] = 0.f; S1[i] = 0.f; }
        {
            v16bf kB = load_frag16(kbase + (size_t)sb * Dm, Dm, lane);
            S0 = __builtin_amdgcn_wmma_f32_16x16x32_bf16(false, rA0, false, kB, (short)0, S0, false, false);
            kB = load_frag16(kbase + (size_t)sb * Dm + 32, Dm, lane);
            S0 = __builtin_amdgcn_wmma_f32_16x16x32_bf16(false, rA1, false, kB, (short)0, S0, false, false);
            kB = load_frag16(kbase + (size_t)(sb + 16) * Dm, Dm, lane);
            S1 = __builtin_amdgcn_wmma_f32_16x16x32_bf16(false, rA0, false, kB, (short)0, S1, false, false);
            kB = load_frag16(kbase + (size_t)(sb + 16) * Dm + 32, Dm, lane);
            S1 = __builtin_amdgcn_wmma_f32_16x16x32_bf16(false, rA1, false, kB, (short)0, S1, false, false);
        }
        // apply fast/slow decay masks, round to bf16, stage in LDS (row-major 16x32)
        const float Cs0 = Cb[sb + (lane & 15)];
        const float Cs1 = Cb[sb + 16 + (lane & 15)];
        const int sl = lane & 15;
#pragma unroll
        for (int i = 0; i < 8; ++i) {
            int ql = ((lane >> 4) << 3) + i;
            float d0 = fabsf(Cq[i] - Cs0);
            float d1 = fabsf(Cq[i] - Cs1);
            smem[wv][0][ql * 32 + sl]      = f2bf(S0[i] * expf(-d0));
            smem[wv][1][ql * 32 + sl]      = f2bf(S0[i] * expf(-d0 * sh));
            smem[wv][0][ql * 32 + 16 + sl] = f2bf(S1[i] * expf(-d1));
            smem[wv][1][ql * 32 + 16 + sl] = f2bf(S1[i] * expf(-d1 * sh));
        }
        v16bf Af = load_frag16(&smem[wv][0][0], 32, lane);
        v16bf As = load_frag16(&smem[wv][1][0], 32, lane);
        // PV: y[16q x 16n] += Smask[16q x 32s] * V[32s x 16n], 4 n-tiles share the Af/As frags
#pragma unroll
        for (int j = 0; j < 4; ++j) {
            v16bf vB = load_frag16(vbase + (size_t)(j * 16) * Tc + sb, Tc, lane);
            yf[j] = __builtin_amdgcn_wmma_f32_16x16x32_bf16(false, Af, false, vB, (short)0, yf[j], false, false);
            ys[j] = __builtin_amdgcn_wmma_f32_16x16x32_bf16(false, As, false, vB, (short)0, ys[j], false, false);
        }
    }
    // alpha mix and store
#pragma unroll
    for (int j = 0; j < 4; ++j) {
        int n = j * 16 + (lane & 15);
        float alpha = 1.f / (1.f + expf(-decay_mix[h * Kc + n]));
#pragma unroll
        for (int i = 0; i < 8; ++i) {
            int q = qt * 16 + ((lane >> 4) << 3) + i;
            y[(size_t)(b * Tc + q) * Dm + h * Kc + n] =
                alpha * yf[j][i] + (1.f - alpha) * ys[j][i];
        }
    }
}

// ---------------- kernel 7: group norm + silu gate ----------------
__global__ void rwkv_gnorm_kernel(const float* __restrict__ y, const float* __restrict__ g,
                                  const float* __restrict__ gamma, const float* __restrict__ beta,
                                  __bf16* __restrict__ yg) {
    int gid = blockIdx.x * blockDim.x + threadIdx.x;
    if (gid >= BT * Hc) return;
    int tok = gid >> 4, h = gid & 15;
    const float* p = y + (size_t)tok * Dm + h * Kc;
    float mu = 0.f;
    for (int i = 0; i < Kc; ++i) mu += p[i];
    mu *= (1.f / Kc);
    float s2 = 0.f;
    for (int i = 0; i < Kc; ++i) { float d = p[i] - mu; s2 += d * d; }
    float rs = rsqrtf(s2 * (1.f / Kc) + EPSc);
    for (int i = 0; i < Kc; ++i) {
        float xn = (p[i] - mu) * rs * gamma[h * Kc + i] + beta[h * Kc + i];
        yg[(size_t)tok * Dm + h * Kc + i] = f2bf(xn * g[(size_t)tok * Dm + h * Kc + i]);
    }
}

// ---------------- host launcher ----------------
extern "C" void kernel_launch(void* const* d_in, const int* in_sizes, int n_in,
                              void* d_out, int out_size, void* d_ws, size_t ws_size,
                              hipStream_t stream) {
    const float* x      = (const float*)d_in[0];
    const float* tmx    = (const float*)d_in[1];
    const float* tmw    = (const float*)d_in[2];
    const float* tmk    = (const float*)d_in[3];
    const float* tmv    = (const float*)d_in[4];
    const float* tmr    = (const float*)d_in[5];
    const float* tmg    = (const float*)d_in[6];
    const float* w1     = (const float*)d_in[7];
    const float* w2     = (const float*)d_in[8];
    const float* tdecay = (const float*)d_in[9];
    const float* dw1    = (const float*)d_in[10];
    const float* dw2    = (const float*)d_in[11];
    const float* hdb    = (const float*)d_in[12];
    const float* sscale = (const float*)d_in[13];
    const float* dmix   = (const float*)d_in[14];
    const float* W_r    = (const float*)d_in[15];
    const float* W_k    = (const float*)d_in[16];
    const float* W_v    = (const float*)d_in[17];
    const float* W_g    = (const float*)d_in[18];
    const float* W_o    = (const float*)d_in[19];
    const float* gamma  = (const float*)d_in[20];
    const float* beta   = (const float*)d_in[21];
    float* out = (float*)d_out;

    // workspace carve-out
    char* wp = (char*)d_ws;
    auto alloc = [&](size_t bytes) -> char* {
        char* p = wp; wp += (bytes + 255) & ~(size_t)255; return p;
    };
    const size_t nbt = (size_t)BT * Dm;
    __bf16* xk_bf = (__bf16*)alloc(nbt * 2);
    __bf16* xv_bf = (__bf16*)alloc(nbt * 2);
    __bf16* xr_bf = (__bf16*)alloc(nbt * 2);
    __bf16* xg_bf = (__bf16*)alloc(nbt * 2);
    __bf16* Wt_r  = (__bf16*)alloc((size_t)Dm * Dm * 2);
    __bf16* Wt_k  = (__bf16*)alloc((size_t)Dm * Dm * 2);
    __bf16* Wt_v  = (__bf16*)alloc((size_t)Dm * Dm * 2);
    __bf16* Wt_g  = (__bf16*)alloc((size_t)Dm * Dm * 2);
    __bf16* Wt_o  = (__bf16*)alloc((size_t)Dm * Dm * 2);
    __bf16* r_bf  = (__bf16*)alloc(nbt * 2);
    __bf16* k_bf  = (__bf16*)alloc(nbt * 2);
    __bf16* v_bf  = (__bf16*)alloc(nbt * 2);
    __bf16* v_t   = (__bf16*)alloc(nbt * 2);
    float*  g_f   = (float*)alloc(nbt * 4);
    float*  wrow  = (float*)alloc((size_t)BT * Hc * 4);
    float*  Cbuf  = (float*)alloc((size_t)Bc * Hc * Tc * 4);
    float*  y_f   = (float*)alloc(nbt * 4);
    __bf16* yg_bf = (__bf16*)alloc(nbt * 2);

    // 1. weight transpose/convert (5 matrices)
    {
        dim3 g((Dm * Dm + 255) / 256), blk(256);
        rwkv_wtrans_kernel<<<g, blk, 0, stream>>>(W_r, Wt_r, Dm, Dm);
        rwkv_wtrans_kernel<<<g, blk, 0, stream>>>(W_k, Wt_k, Dm, Dm);
        rwkv_wtrans_kernel<<<g, blk, 0, stream>>>(W_v, Wt_v, Dm, Dm);
        rwkv_wtrans_kernel<<<g, blk, 0, stream>>>(W_g, Wt_g, Dm, Dm);
        rwkv_wtrans_kernel<<<g, blk, 0, stream>>>(W_o, Wt_o, Dm, Dm);
    }
    // 2. token shift + LoRA mixes + decay row
    rwkv_mix_kernel<<<dim3(BT), dim3(256), 0, stream>>>(
        x, tmx, tmw, tmk, tmv, tmr, tmg, w1, w2, tdecay, dw1, dw2, hdb,
        xk_bf, xv_bf, xr_bf, xg_bf, wrow);
    // 3. per-(b,h) cumsum
    rwkv_cumsum_kernel<<<dim3(Bc * Hc), dim3(32), 0, stream>>>(wrow, Cbuf);
    // 4. projections (WMMA + TDM double buffering); fold K^-0.5 into r
    {
        dim3 g(BT / 64, Dm / 64), blk(128);
        rwkv_gemm_kernel<<<g, blk, 0, stream>>>(xr_bf, Wt_r, r_bf, BT, Dm, Dm, 0, 0.125f);
        rwkv_gemm_kernel<<<g, blk, 0, stream>>>(xk_bf, Wt_k, k_bf, BT, Dm, Dm, 0, 1.0f);
        rwkv_gemm_kernel<<<g, blk, 0, stream>>>(xv_bf, Wt_v, v_bf, BT, Dm, Dm, 0, 1.0f);
        rwkv_gemm_kernel<<<g, blk, 0, stream>>>(xg_bf, Wt_g, g_f,  BT, Dm, Dm, 1, 1.0f);
    }
    // 5. V transpose
    rwkv_vtrans_kernel<<<dim3((BT * Dm + 255) / 256), dim3(256), 0, stream>>>(v_bf, v_t);
    // 6. fused LION attention (WMMA)
    rwkv_attn_kernel<<<dim3(Tc / 64, Hc, Bc), dim3(128), 0, stream>>>(
        r_bf, k_bf, v_t, Cbuf, sscale, dmix, y_f);
    // 7. group norm + gate
    rwkv_gnorm_kernel<<<dim3((BT * Hc + 255) / 256), dim3(256), 0, stream>>>(
        y_f, g_f, gamma, beta, yg_bf);
    // 8. output projection (WMMA + TDM) -> d_out f32
    rwkv_gemm_kernel<<<dim3(BT / 64, Dm / 64), dim3(128), 0, stream>>>(
        yg_bf, Wt_o, out, BT, Dm, Dm, 2, 1.0f);
    (void)in_sizes; (void)n_in; (void)out_size; (void)ws_size;
}
